// ExportSparseMoE_63324997812735
// MI455X (gfx1250) — compile-verified, matrix operand order checked by reading, not run on hardware
//
#include <hip/hip_runtime.h>
#include <hip/hip_bf16.h>
#include <math.h>

// ---------------- problem constants ----------------
#define NE   8      // experts
#define TOPK 2
#define CD   2048   // n_embd
#define HD   1024   // expert intermediate
#define HSD  2048   // shared intermediate
#define NT   512    // tokens (B*T)

typedef __attribute__((ext_vector_type(16))) __bf16 v16bf;
typedef __attribute__((ext_vector_type(2)))  __bf16 v2bf;
typedef __attribute__((ext_vector_type(8)))  float  v8f;
typedef int v4i __attribute__((vector_size(16)));   // matches builtin param pointee

#define NWAVES 8            // 256-thread blocks for the WMMA kernels

// Async global->LDS staging available? (gfx1250 device pass only)
#if defined(__HIP_DEVICE_COMPILE__) && __has_builtin(__builtin_amdgcn_global_load_async_to_lds_b128)
#define HAVE_ASYNC_LDS 1
#else
#define HAVE_ASYNC_LDS 0
#endif

// gateup dynamic LDS: 2 x [16][CD] bf16 tiles + 16KB fp32 async staging chunk
#define GATEUP_SMEM (2 * 16 * CD * 2 + 16384)

// ---------------- bf16 helpers (round-to-nearest-even) ----------------
__device__ __forceinline__ unsigned short f2bf_bits(float f) {
    unsigned u = __float_as_uint(f);
    unsigned r = (u + 0x7FFFu + ((u >> 16) & 1u)) >> 16;
    return (unsigned short)r;
}
__device__ __forceinline__ unsigned pack2bf(float a, float b) {
#if defined(__HIP_DEVICE_COMPILE__) && __has_builtin(__builtin_amdgcn_cvt_pk_bf16_f32)
    union { v2bf v; unsigned u; } o;
    o.v = __builtin_amdgcn_cvt_pk_bf16_f32(a, b);   // v_cvt_pk_bf16_f32
    return o.u;
#else
    return (unsigned)f2bf_bits(a) | ((unsigned)f2bf_bits(b) << 16);
#endif
}
__device__ __forceinline__ __bf16 f2bf(float f) {
    union { unsigned u; __bf16 b[2]; } o; o.u = pack2bf(f, 0.0f); return o.b[0];
}

// ---------------- workspace layout (bytes) ----------------
#define RACC_OFF 0ul                           // float[NT*CD]          4 MiB
#define CNT_OFF  (RACC_OFF + (size_t)NT*CD*4)  // int[NE]
#define TOK_OFF  (CNT_OFF + 256ul)             // int[NE*NT]
#define LW_OFF   (TOK_OFF + (size_t)NE*NT*4)   // float[NE*NT]
#define SG_OFF   (LW_OFF + (size_t)NE*NT*4)    // float[NT]
#define HR_OFF   (SG_OFF + 4096ul)             // bf16[NE*NT*HD]        8 MiB
#define HS_OFF   (HR_OFF + (size_t)NE*NT*HD*2) // bf16[NT*HSD]          2 MiB
#define XBF_OFF  (HS_OFF + (size_t)NT*HSD*2)   // bf16[NT*CD]           2 MiB

// ======================================================================
// Kernel 0: zero routed accumulator + counters, convert x -> bf16 once.
// ======================================================================
__global__ __launch_bounds__(256) void k_prolog(float* __restrict__ racc, int n,
                                                int* __restrict__ cnt,
                                                const float* __restrict__ x,
                                                __bf16* __restrict__ xbf) {
    int i = blockIdx.x * blockDim.x + threadIdx.x;
    if (i < n) racc[i] = 0.0f;
    if (i < NE) cnt[i] = 0;
    if (i < (NT * CD) / 4) {
        float4 v = ((const float4*)x)[i];
        uint2 p; p.x = pack2bf(v.x, v.y); p.y = pack2bf(v.z, v.w);
        ((uint2*)xbf)[i] = p;
    }
}

// ======================================================================
// Kernel 1: routing (scores, top-2 softmax, shared-expert sigmoid gate)
// ======================================================================
__global__ __launch_bounds__(256) void k_route(const float* __restrict__ x,
                                               const float* __restrict__ gw,
                                               const float* __restrict__ segw,
                                               int* __restrict__ cnt,
                                               int* __restrict__ tok,
                                               float* __restrict__ lw,
                                               float* __restrict__ sg) {
    int t = blockIdx.x * blockDim.x + threadIdx.x;
    if (t >= NT) return;
    float acc[NE]; float se = 0.0f;
#pragma unroll
    for (int e = 0; e < NE; ++e) acc[e] = 0.0f;
    const float* xr = x + (size_t)t * CD;
    for (int c = 0; c < CD; c += 4) {
        float4 xv = *(const float4*)(xr + c);
#pragma unroll
        for (int e = 0; e < NE; ++e) {
            const float* g = gw + (size_t)e * CD + c;
            acc[e] += xv.x * g[0] + xv.y * g[1] + xv.z * g[2] + xv.w * g[3];
        }
        const float* s = segw + c;
        se += xv.x * s[0] + xv.y * s[1] + xv.z * s[2] + xv.w * s[3];
    }
    // top-2 (ties -> lower index, matching jax top_k)
    int e0 = 0; float s0 = acc[0];
#pragma unroll
    for (int e = 1; e < NE; ++e) if (acc[e] > s0) { s0 = acc[e]; e0 = e; }
    int e1 = -1; float s1 = -3.0e38f;
#pragma unroll
    for (int e = 0; e < NE; ++e) if (e != e0 && acc[e] > s1) { s1 = acc[e]; e1 = e; }
    float w0 = 1.0f / (1.0f + expf(s1 - s0));   // softmax over {s0,s1}
    float w1 = 1.0f - w0;
    sg[t] = 1.0f / (1.0f + expf(-se));
    int p0 = atomicAdd(&cnt[e0], 1); tok[e0 * NT + p0] = t; lw[e0 * NT + p0] = w0;
    int p1 = atomicAdd(&cnt[e1], 1); tok[e1 * NT + p1] = t; lw[e1 * NT + p1] = w1;
}

// ======================================================================
// Cooperative staging of a contiguous [16][CD] fp32 weight block into an
// LDS bf16 tile. Async (GLOBAL_LOAD_ASYNC_TO_LDS_B128 -> staging chunk ->
// v_cvt_pk_bf16_f32) when available, else direct load+convert.
// tid in [0,256). Caller must __syncthreads() after the final matrix.
// ======================================================================
__device__ __forceinline__ void stage_weights_bf16(const float* __restrict__ src,
                                                   __bf16* __restrict__ dst,
                                                   float* __restrict__ ldsX,
                                                   int tid) {
#if HAVE_ASYNC_LDS
    // 16*CD floats = 8 chunks of 4096 floats (16 KiB staging region).
    // INST_OFFSET applies to BOTH global and LDS addresses (ISA 08 §4.4),
    // so the same immediate advances source and staging chunk in lockstep.
    for (int ch = 0; ch < 8; ++ch) {
        const float* gs = src + ch * 4096 + tid * 4;       // this thread's float4
        float*       ls = ldsX + tid * 4;
        auto g = (__attribute__((address_space(1))) v4i*)(void*)gs;
        auto l = (__attribute__((address_space(3))) v4i*)(void*)ls;
        __builtin_amdgcn_global_load_async_to_lds_b128(g, l, 0, 0);
        __builtin_amdgcn_global_load_async_to_lds_b128(g, l, 4096, 0);
        __builtin_amdgcn_global_load_async_to_lds_b128(g, l, 8192, 0);
        __builtin_amdgcn_global_load_async_to_lds_b128(g, l, 12288, 0);
#if __has_builtin(__builtin_amdgcn_s_wait_asynccnt)
        __builtin_amdgcn_s_wait_asynccnt(0);
#else
        asm volatile("s_wait_asynccnt 0" ::: "memory");
#endif
        __syncthreads();
#pragma unroll
        for (int j = 0; j < 4; ++j) {
            float4 v = ((const float4*)ldsX)[tid + 256 * j];
            uint2 p; p.x = pack2bf(v.x, v.y); p.y = pack2bf(v.z, v.w);
            ((uint2*)dst)[ch * 1024 + tid + 256 * j] = p;
        }
        __syncthreads();
    }
#else
    (void)ldsX;
    for (int i = tid; i < (16 * CD) / 4; i += 256) {
        float4 v = ((const float4*)src)[i];
        uint2 p; p.x = pack2bf(v.x, v.y); p.y = pack2bf(v.z, v.w);
        ((uint2*)dst)[i] = p;
    }
#endif
}

// ======================================================================
// Kernel 2: fused gate/up SwiGLU front half (templated M-blocking).
//   Workgroup = (matrix instance e, 16-row h tile). Weight rows staged
//   once in LDS as bf16; 8 waves sweep token tiles with WMMA bf16.
//   MBLK=2: one B ds_load pair feeds 4 WMMAs (2 token tiles x gate/up).
// ======================================================================
template <int MBLK>
__global__ __launch_bounds__(256) void k_gateup(const __bf16* __restrict__ X,
                                                const float* __restrict__ Wg_base,
                                                const float* __restrict__ Wu_base,
                                                long  wstride,   // per-expert elems (H*C) or 0
                                                int   htiles,    // h tiles per matrix
                                                const int* __restrict__ cnt,  // null => NT
                                                const int* __restrict__ tok,  // null => identity
                                                __bf16* __restrict__ hout,
                                                long  hstride,   // per-expert elems in hout
                                                int   hld) {
    extern __shared__ __align__(16) unsigned char smem[];
    __bf16* ldsG = (__bf16*)smem;            // [16][CD] bf16
    __bf16* ldsU = ldsG + 16 * CD;           // [16][CD] bf16
    float*  ldsX = (float*)(ldsU + 16 * CD); // 16 KiB async staging chunk

    const int e  = blockIdx.x / htiles;
    const int ht = blockIdx.x % htiles;
    const int tid = threadIdx.x;

    const float* wg = Wg_base + (size_t)e * wstride + (size_t)ht * 16 * CD;
    const float* wu = Wu_base + (size_t)e * wstride + (size_t)ht * 16 * CD;
    stage_weights_bf16(wg, ldsG, ldsX, tid);
    stage_weights_bf16(wu, ldsU, ldsX, tid);
    __syncthreads();

    const int ntok   = cnt ? cnt[e] : NT;
    const int ntiles = (ntok + 15) >> 4;
    const int lane   = tid & 31;             // wave32
    const int wave   = tid >> 5;
    const int l15    = lane & 15;
    const int kb     = (lane < 16) ? 0 : 16; // contiguous K half per lane half (same order for A and B)

    for (int tb = wave * MBLK; tb < ntiles; tb += NWAVES * MBLK) {
        // A operand rows for sub-tile 0 (and 1 when MBLK==2)
        int r0    = tb * 16 + l15;
        int ridx0 = (r0 < ntok) ? r0 : (ntok - 1);
        int tk0   = tok ? tok[e * NT + ridx0] : ridx0;
        const __bf16* arow0 = X + (size_t)tk0 * CD + kb;
        const __bf16* arow1 = arow0;
        if constexpr (MBLK > 1) {
            int t1    = (tb + 1 < ntiles) ? tb + 1 : tb;
            int r1    = t1 * 16 + l15;
            int ridx1 = (r1 < ntok) ? r1 : (ntok - 1);
            int tk1   = tok ? tok[e * NT + ridx1] : ridx1;
            arow1 = X + (size_t)tk1 * CD + kb;
        }

        v8f accg0 = {}, accu0 = {}, accg1 = {}, accu1 = {};
        for (int k0 = 0; k0 < CD; k0 += 32) {
            __builtin_prefetch(arow0 + k0 + 128);          // global_prefetch_b8
            v16bf bg = *(const v16bf*)(ldsG + (size_t)l15 * CD + k0 + kb);
            v16bf bu = *(const v16bf*)(ldsU + (size_t)l15 * CD + k0 + kb);
            v16bf a0 = *(const v16bf*)(arow0 + k0);
            accg0 = __builtin_amdgcn_wmma_f32_16x16x32_bf16(false, a0, false, bg,
                                                            (short)0, accg0, false, false);
            accu0 = __builtin_amdgcn_wmma_f32_16x16x32_bf16(false, a0, false, bu,
                                                            (short)0, accu0, false, false);
            if constexpr (MBLK > 1) {
                __builtin_prefetch(arow1 + k0 + 128);
                v16bf a1 = *(const v16bf*)(arow1 + k0);
                accg1 = __builtin_amdgcn_wmma_f32_16x16x32_bf16(false, a1, false, bg,
                                                                (short)0, accg1, false, false);
                accu1 = __builtin_amdgcn_wmma_f32_16x16x32_bf16(false, a1, false, bu,
                                                                (short)0, accu1, false, false);
            }
        }
        // epilogue: h = silu(g) * u, store bf16
        const int hcol = ht * 16 + l15;                      // N = lane%16
        const int half = (lane >= 16) ? 8 : 0;               // C/D layout M split
        __bf16* ho = hout + (size_t)e * hstride;
#pragma unroll
        for (int v = 0; v < 8; ++v) {
            int mrow = tb * 16 + v + half;
            float g = accg0[v], u = accu0[v];
            float h = (g / (1.0f + expf(-g))) * u;
            ho[(size_t)mrow * hld + hcol] = f2bf(h);
        }
        if constexpr (MBLK > 1) {
            if (tb + 1 < ntiles) {
#pragma unroll
                for (int v = 0; v < 8; ++v) {
                    int mrow = (tb + 1) * 16 + v + half;
                    float g = accg1[v], u = accu1[v];
                    float h = (g / (1.0f + expf(-g))) * u;
                    ho[(size_t)mrow * hld + hcol] = f2bf(h);
                }
            }
        }
    }
}

// ======================================================================
// Kernel 3: routed down-projection, scaled by routing weight, atomic
//   accumulation into racc. Workgroup = (expert, 16-col c tile).
// ======================================================================
__global__ __launch_bounds__(256) void k_down_routed(const __bf16* __restrict__ hbuf,
                                                     const float* __restrict__ wdown,
                                                     const int* __restrict__ cnt,
                                                     const int* __restrict__ tok,
                                                     const float* __restrict__ lw,
                                                     float* __restrict__ racc) {
    extern __shared__ __align__(16) unsigned char smem[];
    __bf16* ldsW = (__bf16*)smem;            // [16][HD] bf16 (32 KiB)

    const int ctiles = CD / 16;
    const int e  = blockIdx.x / ctiles;
    const int ct = blockIdx.x % ctiles;
    const int tid = threadIdx.x;

    const float* wd = wdown + (size_t)e * CD * HD + (size_t)ct * 16 * HD;
    for (int i = tid; i < (16 * HD) / 4; i += 256) {
        float4 w = ((const float4*)wd)[i];
        uint2 p; p.x = pack2bf(w.x, w.y); p.y = pack2bf(w.z, w.w);
        ((uint2*)ldsW)[i] = p;
    }
    __syncthreads();

    const int ntok   = cnt[e];
    const int ntiles = (ntok + 15) >> 4;
    const int lane   = tid & 31;
    const int wave   = tid >> 5;
    const int l15    = lane & 15;
    const int kb     = (lane < 16) ? 0 : 16;
    const __bf16* hb = hbuf + (size_t)e * NT * HD;

    for (int tile = wave; tile < ntiles; tile += NWAVES) {
        const __bf16* arow = hb + (size_t)(tile * 16 + l15) * HD + kb;
        v8f acc = {};
        for (int k0 = 0; k0 < HD; k0 += 32) {
            __builtin_prefetch(arow + k0 + 128);
            v16bf a = *(const v16bf*)(arow + k0);
            v16bf b = *(const v16bf*)(ldsW + (size_t)l15 * HD + k0 + kb);
            acc = __builtin_amdgcn_wmma_f32_16x16x32_bf16(false, a, false, b,
                                                          (short)0, acc, false, false);
        }
        const int ccol = ct * 16 + l15;
        const int half = (lane >= 16) ? 8 : 0;
#pragma unroll
        for (int v = 0; v < 8; ++v) {
            int pos = tile * 16 + v + half;
            if (pos < ntok) {
                int   t = tok[e * NT + pos];
                float w = lw[e * NT + pos];
                atomicAdd(&racc[(size_t)t * CD + ccol], w * acc[v]);  // global_atomic_add_f32
            }
        }
    }
}

// ======================================================================
// Kernel 4: shared-expert down-projection + final combine:
//   out = racc + sigmoid(se_gate) * shared_down.
//   2-tile M-blocking: one B ds_load pair feeds 2 WMMAs.
// ======================================================================
__global__ __launch_bounds__(256) void k_down_shared(const __bf16* __restrict__ hsh,
                                                     const float* __restrict__ shdown,
                                                     const float* __restrict__ racc,
                                                     const float* __restrict__ sg,
                                                     float* __restrict__ out) {
    extern __shared__ __align__(16) unsigned char smem[];
    __bf16* ldsW = (__bf16*)smem;            // [16][HSD] bf16 (64 KiB)

    const int ct  = blockIdx.x;              // grid = CD/16
    const int tid = threadIdx.x;

    const float* wd = shdown + (size_t)ct * 16 * HSD;
    for (int i = tid; i < (16 * HSD) / 4; i += 256) {
        float4 w = ((const float4*)wd)[i];
        uint2 p; p.x = pack2bf(w.x, w.y); p.y = pack2bf(w.z, w.w);
        ((uint2*)ldsW)[i] = p;
    }
    __syncthreads();

    const int lane = tid & 31;
    const int wave = tid >> 5;
    const int l15  = lane & 15;
    const int kb   = (lane < 16) ? 0 : 16;

    // NT/16 = 32 tiles, processed in pairs: tb = 2*wave, 2*wave+16
    for (int tb = wave * 2; tb < NT / 16; tb += NWAVES * 2) {
        const __bf16* arow0 = hsh + (size_t)(tb * 16 + l15) * HSD + kb;
        const __bf16* arow1 = hsh + (size_t)((tb + 1) * 16 + l15) * HSD + kb;
        v8f acc0 = {}, acc1 = {};
        for (int k0 = 0; k0 < HSD; k0 += 32) {
            __builtin_prefetch(arow0 + k0 + 128);
            __builtin_prefetch(arow1 + k0 + 128);
            v16bf b  = *(const v16bf*)(ldsW + (size_t)l15 * HSD + k0 + kb);
            v16bf a0 = *(const v16bf*)(arow0 + k0);
            v16bf a1 = *(const v16bf*)(arow1 + k0);
            acc0 = __builtin_amdgcn_wmma_f32_16x16x32_bf16(false, a0, false, b,
                                                           (short)0, acc0, false, false);
            acc1 = __builtin_amdgcn_wmma_f32_16x16x32_bf16(false, a1, false, b,
                                                           (short)0, acc1, false, false);
        }
        const int ccol = ct * 16 + l15;
        const int half = (lane >= 16) ? 8 : 0;
#pragma unroll
        for (int v = 0; v < 8; ++v) {
            int t0 = tb * 16 + v + half;
            int t1 = (tb + 1) * 16 + v + half;
            out[(size_t)t0 * CD + ccol] = racc[(size_t)t0 * CD + ccol] + sg[t0] * acc0[v];
            out[(size_t)t1 * CD + ccol] = racc[(size_t)t1 * CD + ccol] + sg[t1] * acc1[v];
        }
    }
}

// ======================================================================
extern "C" void kernel_launch(void* const* d_in, const int* in_sizes, int n_in,
                              void* d_out, int out_size, void* d_ws, size_t ws_size,
                              hipStream_t stream) {
    const float* x        = (const float*)d_in[0];
    const float* gate_w   = (const float*)d_in[1];
    const float* w_gate   = (const float*)d_in[2];
    const float* w_up     = (const float*)d_in[3];
    const float* w_down   = (const float*)d_in[4];
    const float* sh_gate  = (const float*)d_in[5];
    const float* sh_up    = (const float*)d_in[6];
    const float* sh_down  = (const float*)d_in[7];
    const float* se_gate  = (const float*)d_in[8];
    float* out = (float*)d_out;

    char* ws = (char*)d_ws;
    float*  racc = (float*)(ws + RACC_OFF);
    int*    cnt  = (int*)  (ws + CNT_OFF);
    int*    tok  = (int*)  (ws + TOK_OFF);
    float*  lw   = (float*)(ws + LW_OFF);
    float*  sg   = (float*)(ws + SG_OFF);
    __bf16* hrt  = (__bf16*)(ws + HR_OFF);
    __bf16* hsh  = (__bf16*)(ws + HS_OFF);
    __bf16* xbf  = (__bf16*)(ws + XBF_OFF);

    // 0) zero accumulator + counters, convert activations to bf16
    k_prolog<<<(NT * CD + 255) / 256, 256, 0, stream>>>(racc, NT * CD, cnt, x, xbf);
    // 1) routing
    k_route<<<(NT + 255) / 256, 256, 0, stream>>>(x, gate_w, se_gate, cnt, tok, lw, sg);
    // 2a) routed experts: gate/up SwiGLU   (grid = 8 experts x 64 h-tiles)
    k_gateup<1><<<NE * (HD / 16), 256, GATEUP_SMEM, stream>>>(
        xbf, w_gate, w_up, (long)HD * CD, HD / 16, cnt, tok,
        hrt, (long)NT * HD, HD);
    // 2b) shared expert: gate/up SwiGLU    (grid = 128 h-tiles, 2-tile blocking)
    k_gateup<2><<<HSD / 16, 256, GATEUP_SMEM, stream>>>(
        xbf, sh_gate, sh_up, 0L, HSD / 16, nullptr, nullptr,
        hsh, 0L, HSD);
    // 3) routed down-projection, weighted atomic accumulate
    k_down_routed<<<NE * (CD / 16), 256, 16 * HD * 2, stream>>>(
        hrt, w_down, cnt, tok, lw, racc);
    // 4) shared down-projection + final combine
    k_down_shared<<<CD / 16, 256, 16 * HSD * 2, stream>>>(
        hsh, sh_down, racc, sg, out);
}
